// MobiusGRU_52432960749616
// MI455X (gfx1250) — compile-verified, matrix operand order checked by reading, not run on hardware
//
#include <hip/hip_runtime.h>

// ---------------------------------------------------------------------------
// MI455X / gfx1250  Mobius (hyperbolic) bidirectional GRU
//   - bf16 WMMA 16x16x32, f32 accumulate for all matmuls
//   - TDM (tensor_load_to_lds) async staging of A-tiles, s_wait_tensorcnt
//   - fused GEMM + hyperbolic epilogue, one wave per batch row (wave32)
// ---------------------------------------------------------------------------

typedef __attribute__((ext_vector_type(16))) __bf16 v16bf;
typedef __attribute__((ext_vector_type(8)))  float  v8f;
typedef __attribute__((ext_vector_type(4)))  unsigned int u32x4;
typedef __attribute__((ext_vector_type(8)))  int i32x8;
typedef __attribute__((ext_vector_type(4)))  int i32x4;

#define TT 256
#define BB 128
#define DD 512
#define HH 512

#ifndef __has_builtin
#define __has_builtin(x) 0
#endif
#if __has_builtin(__builtin_amdgcn_tensor_load_to_lds) && \
    __has_builtin(__builtin_amdgcn_s_wait_tensorcnt)
#define HAVE_TDM 1
#else
#define HAVE_TDM 0
#endif

// ---------------- scalar helpers ----------------
__device__ __forceinline__ unsigned short f2bf(float f) {
  unsigned u = __builtin_bit_cast(unsigned, f);
  return (unsigned short)((u + 0x7FFFu + ((u >> 16) & 1u)) >> 16);
}
__device__ __forceinline__ float bf2f(unsigned short s) {
  unsigned u = ((unsigned)s) << 16;
  return __builtin_bit_cast(float, u);
}
// wave32 all-reduce (sum)
__device__ __forceinline__ float wsum(float v) {
  #pragma unroll
  for (int m = 16; m >= 1; m >>= 1) v += __shfl_xor(v, m, 32);
  return v;
}
__device__ __forceinline__ float artanh_c(float x) {
  x = fminf(fmaxf(x, -1.f + 1e-7f), 1.f - 1e-7f);
  return 0.5f * logf((1.f + x) / (1.f - x));
}

// ---------------- row-vector helpers (16 f32 per lane == 512 per wave) -----
__device__ __forceinline__ float norm16(const float v[16]) {
  float s = 0.f;
  #pragma unroll
  for (int j = 0; j < 16; ++j) s += v[j] * v[j];
  return fmaxf(sqrtf(wsum(s)), 1e-15f);
}
__device__ __forceinline__ void mobius_add16(const float x[16], const float y[16], float o[16]) {
  float dxy = 0.f, dx2 = 0.f, dy2 = 0.f;
  #pragma unroll
  for (int j = 0; j < 16; ++j) { dxy += x[j]*y[j]; dx2 += x[j]*x[j]; dy2 += y[j]*y[j]; }
  dxy = wsum(dxy); dx2 = wsum(dx2); dy2 = wsum(dy2);
  const float cx  = 1.f + 2.f*dxy + dy2;
  const float cy  = 1.f - dx2;
  const float den = fmaxf(1.f + 2.f*dxy + dx2*dy2, 1e-15f);
  #pragma unroll
  for (int j = 0; j < 16; ++j) o[j] = (cx*x[j] + cy*y[j]) / den;
}

// ---------------- TDM async staging of one contiguous 16KB bf16 tile -------
// The 16 adjacent batch rows of h / x / rh are contiguous in memory, so the
// tile is a flat 8192-element (2B) 1-D tensor: trivial D# per ISA 08 §8.3/8.4.
__device__ __forceinline__ void stage_tile(unsigned short* sDst,
                                           const unsigned short* gSrc, int tid) {
#if HAVE_TDM
  if (tid < 32) {                                  // wave 0 issues the DMA
    const unsigned lds_off = (unsigned)(unsigned long long)(void*)sDst;
    const unsigned long long ga = (unsigned long long)(const void*)gSrc;
    u32x4 g0;
    g0[0] = 1u;                                    // count=1, user descriptor
    g0[1] = lds_off;                               // lds_addr (bytes)
    g0[2] = (unsigned)(ga & 0xFFFFFFFFu);          // global_addr[31:0]
    g0[3] = (unsigned)((ga >> 32) & 0x01FFFFFFu)   // global_addr[56:32]
            | 0x80000000u;                         // type=2 ("image")
    i32x8 g1 = {};
    g1[0] = 0x00010000;                            // data_size=1 -> 2 bytes
    g1[1] = 0x20000000;                            // tensor_dim0[15:0]=8192 @bits63:48
    g1[2] = 0;                                     // tensor_dim0[31:16]=0, tensor_dim1=0
    g1[3] = 0x20000000;                            // tile_dim0=8192 @bits127:112
    g1[4] = 0;                                     // tile_dim1=0 (1-D), tile_dim2=0
    g1[5] = 8192;                                  // tensor_dim0_stride
    g1[6] = 0; g1[7] = 0;
    i32x4 z4 = {};
#if __clang_major__ >= 23
    i32x8 z8 = {};
    __builtin_amdgcn_tensor_load_to_lds(g0, g1, z4, z4, z8, 0);
#else
    __builtin_amdgcn_tensor_load_to_lds(g0, g1, z4, z4, 0);
#endif
  }
#else
  for (int i = tid; i < 16 * DD; i += 512) sDst[i] = gSrc[i];
#endif
}
__device__ __forceinline__ void stage_wait(int tid) {
#if HAVE_TDM
  if (tid < 32) __builtin_amdgcn_s_wait_tensorcnt(0);
#endif
}

// ---------------- WMMA micro-GEMM ------------------------------------------
// Workgroup = 512 threads = 16 wave32.  Each wave owns two 16-col N-tiles
// for TWO result matrices:  A1(16xK) @ W1^T  and  A2(16xK) @ W2^T, K=512.
// A tiles staged in LDS (bf16), B tiles streamed from L2-resident bf16 W.
__device__ __forceinline__ void gemm2(const unsigned short* sA1, const unsigned short* sA2,
                                      const unsigned short* W1, const unsigned short* W2,
                                      int wave, int lane, v8f cH[2], v8f cX[2]) {
  const int m    = lane & 15;
  const int side = lane >> 4;
  const int ko   = side * 16;
  const int nt0  = wave * 2;
  __builtin_prefetch(&W1[(size_t)(nt0 * 16 + m) * DD], 0, 0);   // global_prefetch_b8
  __builtin_prefetch(&W2[(size_t)(nt0 * 16 + m) * DD], 0, 0);
  v8f z = {};
  cH[0] = z; cH[1] = z; cX[0] = z; cX[1] = z;
  for (int k0 = 0; k0 < DD; k0 += 32) {
    v16bf a1 = *(const v16bf*)&sA1[m * DD + k0 + ko];
    v16bf a2 = *(const v16bf*)&sA2[m * DD + k0 + ko];
    #pragma unroll
    for (int t = 0; t < 2; ++t) {
      const size_t n = (size_t)((nt0 + t) * 16 + m);
      v16bf b1 = *(const v16bf*)&W1[n * DD + k0 + ko];
      v16bf b2 = *(const v16bf*)&W2[n * DD + k0 + ko];
      cH[t] = __builtin_amdgcn_wmma_f32_16x16x32_bf16(false, a1, false, b1,
                                                      (short)0, cH[t], false, false);
      cX[t] = __builtin_amdgcn_wmma_f32_16x16x32_bf16(false, a2, false, b2,
                                                      (short)0, cX[t], false, false);
    }
  }
}

// C/D layout: VGPR r, lanes 0-15 -> (M=r, N=lane); lanes 16-31 -> (M=8+r).
__device__ __forceinline__ void storeTiles(const v8f c[2], float* sR, int wave, int lane) {
  const int m = lane & 15, side = lane >> 4, nt0 = wave * 2;
  #pragma unroll
  for (int t = 0; t < 2; ++t)
    #pragma unroll
    for (int r = 0; r < 8; ++r)
      sR[(r + 8 * side) * HH + (nt0 + t) * 16 + m] = c[t][r];
}
__device__ __forceinline__ void loadRow(const float* sR, int wave, int lane, float v[16]) {
  #pragma unroll
  for (int j = 0; j < 16; ++j) v[j] = sR[wave * HH + lane + 32 * j];
}

// ---------------------------------------------------------------------------
// gate_kernel : blockIdx.x = M-tile (16 batch rows), blockIdx.y = gate (0=z,1=r)
// ---------------------------------------------------------------------------
extern "C" __global__ void __launch_bounds__(512)
gate_kernel(const float* __restrict__ hF, const unsigned short* __restrict__ hB,
            const float* __restrict__ xF, const unsigned short* __restrict__ xB,
            const unsigned short* __restrict__ Whh, const unsigned short* __restrict__ Wih,
            const float* __restrict__ bias3, const float* __restrict__ lng3,
            const float* __restrict__ lnb3,
            float* __restrict__ zOut, unsigned short* __restrict__ rhOut) {
  __shared__ __align__(32) unsigned short sH[16 * DD];
  __shared__ __align__(32) unsigned short sX[16 * DD];
  __shared__ float sR[16 * HH];
  const int tid = threadIdx.x, wave = tid >> 5, lane = tid & 31;
  const int m0 = blockIdx.x * 16;
  const int g  = blockIdx.y;                 // 0 = z, 1 = r

  stage_tile(sH, hB + (size_t)m0 * HH, tid);
  stage_tile(sX, xB + (size_t)m0 * DD, tid);
  stage_wait(tid);
  __syncthreads();

  const int row = m0 + wave;
  float hreg[16]; float hs = 0.f, xs = 0.f;
  #pragma unroll
  for (int j = 0; j < 16; ++j) {
    const int c = lane + 32 * j;
    const float hv = hF[(size_t)row * HH + c]; hreg[j] = hv; hs += hv * hv;
    const float xv = xF[(size_t)row * DD + c]; xs += xv * xv;
  }
  const float hn = fmaxf(sqrtf(wsum(hs)), 1e-15f);
  const float xn = fmaxf(sqrtf(wsum(xs)), 1e-15f);

  // z gate -> W rows [2H,3H), b[2], ln row 0 ; r gate -> rows [0,H), b[0], ln row 1
  const size_t wofs = (g == 0) ? (size_t)2 * HH * DD : 0;
  const float* bg = bias3 + (g == 0 ? 2 * HH : 0);
  const float* gg = lng3  + (g == 0 ? 0 : HH);
  const float* gb = lnb3  + (g == 0 ? 0 : HH);

  v8f cH[2], cX[2];
  gemm2(sH, sX, Whh + wofs, Wih + wofs, wave, lane, cH, cX);
  __syncthreads();
  storeTiles(cH, sR, wave, lane); __syncthreads();
  float mh[16]; loadRow(sR, wave, lane, mh); __syncthreads();
  storeTiles(cX, sR, wave, lane); __syncthreads();
  float mx[16]; loadRow(sR, wave, lane, mx);

  float wm[16], um[16];
  { const float mn = norm16(mh); const float f = tanhf(mn / hn * artanh_c(hn)) / mn;
    #pragma unroll
    for (int j = 0; j < 16; ++j) wm[j] = f * mh[j]; }
  { const float mn = norm16(mx); const float f = tanhf(mn / xn * artanh_c(xn)) / mn;
    #pragma unroll
    for (int j = 0; j < 16; ++j) um[j] = f * mx[j]; }

  float s1[16]; mobius_add16(wm, um, s1);
  float bv[16];
  #pragma unroll
  for (int j = 0; j < 16; ++j) bv[j] = bg[lane + 32 * j];
  float s2[16]; mobius_add16(s1, bv, s2);

  float pre[16];
  { const float sn = norm16(s2); const float f = artanh_c(sn) / sn;
    #pragma unroll
    for (int j = 0; j < 16; ++j) pre[j] = f * s2[j]; }

  float mu = 0.f;
  #pragma unroll
  for (int j = 0; j < 16; ++j) mu += pre[j];
  mu = wsum(mu) * (1.f / HH);
  float dv[16], var = 0.f;
  #pragma unroll
  for (int j = 0; j < 16; ++j) { dv[j] = pre[j] - mu; var += dv[j] * dv[j]; }
  var = wsum(var) * (1.f / HH);
  const float inv = rsqrtf(var + 1e-5f);

  float gate[16];
  #pragma unroll
  for (int j = 0; j < 16; ++j) {
    const int c = lane + 32 * j;
    const float ln = dv[j] * inv * gg[c] + gb[c];
    gate[j] = 1.f / (1.f + expf(-ln));
  }

  if (g == 0) {
    #pragma unroll
    for (int j = 0; j < 16; ++j) zOut[(size_t)row * HH + lane + 32 * j] = gate[j];
  } else {
    float rh[16]; float rs = 0.f;
    #pragma unroll
    for (int j = 0; j < 16; ++j) { rh[j] = gate[j] * hreg[j]; rs += rh[j] * rh[j]; }
    const float rn = fmaxf(sqrtf(wsum(rs)), 1e-15f);
    const float fr = tanhf(rn / hn * artanh_c(hn)) / rn;
    #pragma unroll
    for (int j = 0; j < 16; ++j)
      rhOut[(size_t)row * HH + lane + 32 * j] = f2bf(fr * rh[j]);
  }
}

// ---------------------------------------------------------------------------
// cand_kernel : candidate GEMM on rh + full h-update epilogue
// ---------------------------------------------------------------------------
extern "C" __global__ void __launch_bounds__(512)
cand_kernel(const unsigned short* __restrict__ rhB,
            const float* __restrict__ xF, const unsigned short* __restrict__ xB,
            const unsigned short* __restrict__ WhhH, const unsigned short* __restrict__ WihH,
            const float* __restrict__ bh, const float* __restrict__ gg,
            const float* __restrict__ gb,
            float* __restrict__ hF, unsigned short* __restrict__ hB,
            const float* __restrict__ zW,
            float* __restrict__ outBase, int t, int dir, int isLast,
            float* __restrict__ htOut) {
  __shared__ __align__(32) unsigned short sA[16 * DD];   // rh (bf16)
  __shared__ __align__(32) unsigned short sX[16 * DD];
  __shared__ float sR[16 * HH];
  const int tid = threadIdx.x, wave = tid >> 5, lane = tid & 31;
  const int m0 = blockIdx.x * 16;

  stage_tile(sA, rhB + (size_t)m0 * HH, tid);
  stage_tile(sX, xB + (size_t)m0 * DD, tid);
  stage_wait(tid);
  __syncthreads();

  const int row = m0 + wave;
  float hreg[16], zreg[16], rreg[16];
  float hs = 0.f, xs = 0.f, rs = 0.f;
  #pragma unroll
  for (int j = 0; j < 16; ++j) {
    const int c = lane + 32 * j;
    const float hv = hF[(size_t)row * HH + c]; hreg[j] = hv; hs += hv * hv;
    const float xv = xF[(size_t)row * DD + c]; xs += xv * xv;
    const float rv = bf2f(sA[wave * DD + c]);  rreg[j] = rv; rs += rv * rv;
    zreg[j] = zW[(size_t)row * HH + c];
  }
  const float hn = fmaxf(sqrtf(wsum(hs)), 1e-15f);
  const float xn = fmaxf(sqrtf(wsum(xs)), 1e-15f);
  const float rn = fmaxf(sqrtf(wsum(rs)), 1e-15f);

  v8f cH[2], cX[2];
  gemm2(sA, sX, WhhH, WihH, wave, lane, cH, cX);
  __syncthreads();
  storeTiles(cH, sR, wave, lane); __syncthreads();
  float mh[16]; loadRow(sR, wave, lane, mh); __syncthreads();
  storeTiles(cX, sR, wave, lane); __syncthreads();
  float mx[16]; loadRow(sR, wave, lane, mx);

  float wm[16], um[16];
  { const float mn = norm16(mh); const float f = tanhf(mn / rn * artanh_c(rn)) / mn;
    #pragma unroll
    for (int j = 0; j < 16; ++j) wm[j] = f * mh[j]; }
  { const float mn = norm16(mx); const float f = tanhf(mn / xn * artanh_c(xn)) / mn;
    #pragma unroll
    for (int j = 0; j < 16; ++j) um[j] = f * mx[j]; }

  float s1[16]; mobius_add16(wm, um, s1);
  float bv[16];
  #pragma unroll
  for (int j = 0; j < 16; ++j) bv[j] = bh[lane + 32 * j];
  float s2[16]; mobius_add16(s1, bv, s2);

  float lt[16];
  { const float sn = norm16(s2); const float f = artanh_c(sn) / sn;
    #pragma unroll
    for (int j = 0; j < 16; ++j) lt[j] = f * s2[j]; }

  float mu = 0.f;
  #pragma unroll
  for (int j = 0; j < 16; ++j) mu += lt[j];
  mu = wsum(mu) * (1.f / HH);
  float dv[16], var = 0.f;
  #pragma unroll
  for (int j = 0; j < 16; ++j) { dv[j] = lt[j] - mu; var += dv[j] * dv[j]; }
  var = wsum(var) * (1.f / HH);
  const float inv = rsqrtf(var + 1e-5f);

  float th[16];
  #pragma unroll
  for (int j = 0; j < 16; ++j) {
    const int c = lane + 32 * j;
    th[j] = tanhf(dv[j] * inv * gg[c] + gb[c]);
  }
  float htl[16];                                 // h_tilde = expmap0(th)
  { const float tn = norm16(th); const float f = tanhf(tn) / tn;
    #pragma unroll
    for (int j = 0; j < 16; ++j) htl[j] = f * th[j]; }

  // h_new = madd( map((1-z)*h | h), map(z*h_tilde | h_tilde) )
  float av[16], bw[16];
  #pragma unroll
  for (int j = 0; j < 16; ++j) { av[j] = (1.f - zreg[j]) * hreg[j]; bw[j] = zreg[j] * htl[j]; }
  float am[16], bm[16];
  { const float an = norm16(av); const float f = tanhf(an / hn * artanh_c(hn)) / an;
    #pragma unroll
    for (int j = 0; j < 16; ++j) am[j] = f * av[j]; }
  { const float tn = norm16(htl); const float bn = norm16(bw);
    const float f = tanhf(bn / tn * artanh_c(tn)) / bn;
    #pragma unroll
    for (int j = 0; j < 16; ++j) bm[j] = f * bw[j]; }
  float hnew[16]; mobius_add16(am, bm, hnew);

  const size_t outRow = ((size_t)t * BB + row) * (2 * HH) + (dir == 0 ? HH : 0);
  #pragma unroll
  for (int j = 0; j < 16; ++j) {
    const int c = lane + 32 * j;
    hF[(size_t)row * HH + c] = hnew[j];
    hB[(size_t)row * HH + c] = f2bf(hnew[j]);
    outBase[outRow + c] = hnew[j];
  }
  if (isLast) {
    #pragma unroll
    for (int j = 0; j < 16; ++j) htOut[(size_t)row * HH + lane + 32 * j] = hnew[j];
  }
}

// ---------------------------------------------------------------------------
extern "C" __global__ void __launch_bounds__(512)
project_kernel(float* __restrict__ out) {
  const int wave = threadIdx.x >> 5, lane = threadIdx.x & 31;
  const size_t row = (size_t)blockIdx.x * 16 + wave;
  float v[32]; float s = 0.f;
  #pragma unroll
  for (int j = 0; j < 32; ++j) { v[j] = out[row * 1024 + lane + 32 * j]; s += v[j] * v[j]; }
  const float n = fmaxf(sqrtf(wsum(s)), 1e-15f);
  const float maxn = 1.f - 0.004f;
  const float f = (n > maxn) ? (maxn / n) : 1.f;
  #pragma unroll
  for (int j = 0; j < 32; ++j) out[row * 1024 + lane + 32 * j] = v[j] * f;
}

extern "C" __global__ void cvt_bf16(const float* __restrict__ src,
                                    unsigned short* __restrict__ dst, size_t n) {
  const size_t i = (size_t)blockIdx.x * blockDim.x + threadIdx.x;
  if (i < n) dst[i] = f2bf(src[i]);
}
extern "C" __global__ void zero_h(float* __restrict__ hF,
                                  unsigned short* __restrict__ hB, size_t n) {
  const size_t i = (size_t)blockIdx.x * blockDim.x + threadIdx.x;
  if (i < n) { hF[i] = 0.f; hB[i] = 0; }
}

// ---------------------------------------------------------------------------
extern "C" void kernel_launch(void* const* d_in, const int* in_sizes, int n_in,
                              void* d_out, int out_size, void* d_ws, size_t ws_size,
                              hipStream_t stream) {
  (void)in_sizes; (void)n_in; (void)out_size; (void)ws_size;
  const float* x    = (const float*)d_in[0];
  const float* Wih  = (const float*)d_in[1];
  const float* Whh  = (const float*)d_in[2];
  const float* bias = (const float*)d_in[3];
  const float* lng  = (const float*)d_in[4];
  const float* lnb  = (const float*)d_in[5];
  float* out = (float*)d_out;

  const size_t nX = (size_t)TT * BB * DD;
  const size_t nW = (size_t)2 * 3 * HH * DD;
  const size_t BH = (size_t)BB * HH;

  char* p = (char*)d_ws;
  unsigned short* xbf   = (unsigned short*)p; p += nX * 2;
  unsigned short* wihbf = (unsigned short*)p; p += nW * 2;
  unsigned short* whhbf = (unsigned short*)p; p += nW * 2;
  float*          hF    = (float*)p;          p += 2 * BH * 4;
  unsigned short* hB    = (unsigned short*)p; p += 2 * BH * 2;
  float*          zW    = (float*)p;          p += 2 * BH * 4;
  unsigned short* rhB   = (unsigned short*)p; p += 2 * BH * 2;

  cvt_bf16<<<(unsigned)((nX + 255) / 256), 256, 0, stream>>>(x,   xbf,   nX);
  cvt_bf16<<<(unsigned)((nW + 255) / 256), 256, 0, stream>>>(Wih, wihbf, nW);
  cvt_bf16<<<(unsigned)((nW + 255) / 256), 256, 0, stream>>>(Whh, whhbf, nW);
  zero_h  <<<(unsigned)((2 * BH + 255) / 256), 256, 0, stream>>>(hF, hB, 2 * BH);

  float* htBase = out + (size_t)TT * BB * 2 * HH;      // ht = [h_last_n, h_last_p]

  for (int dir = 0; dir < 2; ++dir) {
    const unsigned short* WihD = wihbf + (size_t)dir * 3 * HH * DD;
    const unsigned short* WhhD = whhbf + (size_t)dir * 3 * HH * DD;
    const float* biasD = bias + (size_t)dir * 3 * HH;
    const float* lngD  = lng  + (size_t)dir * 3 * HH;
    const float* lnbD  = lnb  + (size_t)dir * 3 * HH;
    float* hFd = hF + (size_t)dir * BH;
    unsigned short* hBd = hB + (size_t)dir * BH;
    float* zWd = zW + (size_t)dir * BH;
    unsigned short* rhd = rhB + (size_t)dir * BH;
    float* htD = htBase + (dir == 1 ? 0 : BH);

    for (int s = 0; s < TT; ++s) {
      const int t = dir ? (TT - 1 - s) : s;
      const float* xFt = x + (size_t)t * BB * DD;
      const unsigned short* xBt = xbf + (size_t)t * BB * DD;

      dim3 gGrid(BB / 16, 2);
      gate_kernel<<<gGrid, 512, 0, stream>>>(hFd, hBd, xFt, xBt, WhhD, WihD,
                                             biasD, lngD, lnbD, zWd, rhd);
      cand_kernel<<<BB / 16, 512, 0, stream>>>(rhd, xFt, xBt,
                                               WhhD + (size_t)HH * DD,
                                               WihD + (size_t)HH * DD,
                                               biasD + HH, lngD + 2 * HH, lnbD + 2 * HH,
                                               hFd, hBd, zWd, out, t, dir,
                                               (s == TT - 1) ? 1 : 0, htD);
    }
  }
  project_kernel<<<(TT * BB) / 16, 512, 0, stream>>>(out);
}
